// Self_Attention_layer_14250701488517
// MI455X (gfx1250) — matrix operand order repaired and checked by reading, not compiled, not run on hardware
//
#include <hip/hip_runtime.h>
#include <stdint.h>

// Self-attention: B=8, C=256, H=W=64 (N=4096), fp32 in/out, f16 WMMA internally.

typedef _Float16 half_t;
typedef _Float16 v16h __attribute__((ext_vector_type(16)));
typedef _Float16 v8h  __attribute__((ext_vector_type(8)));
typedef float    v8f  __attribute__((ext_vector_type(8)));
typedef int      v4i  __attribute__((ext_vector_type(4)));

#define BATCH 8
#define CH    256
#define NPIX  4096

// LDS row strides (halfs) — padded to de-phase banks, 16B-aligned rows.
#define KSTRIDE 264   // 32 K rows  x 528 B
#define VSTRIDE 40    // 256 V rows x 80 B

static __device__ __forceinline__ v8f wmma_f16(v16h a, v16h b, v8f c) {
    return __builtin_amdgcn_wmma_f32_16x16x32_f16(false, a, false, b, (short)0, c,
                                                  false, false);
}

static __device__ __forceinline__ v8f zero_v8f() {
    v8f z = {0.f, 0.f, 0.f, 0.f, 0.f, 0.f, 0.f, 0.f};
    return z;
}

// Assemble a 16-half fragment from two 16B chunks (16B-aligned each).
static __device__ __forceinline__ v16h frag_from2(const half_t* p0, const half_t* p1) {
    v8h lo = *(const v8h*)p0;
    v8h hi = *(const v8h*)p1;
    v16h r;
    #pragma unroll
    for (int e = 0; e < 8; ++e) { r[e] = lo[e]; r[8 + e] = hi[e]; }
    return r;
}

// ---------------------------------------------------------------------------
// Async global -> LDS copy (CDNA5 GLOBAL_LOAD_ASYNC_TO_LDS_B128), with a
// synchronous load+ds_store fallback if the builtin is unavailable.
// Probe-confirmed: param 0 is addrspace(1) int4* (global source).
// ---------------------------------------------------------------------------
#if __has_builtin(__builtin_amdgcn_global_load_async_to_lds_b128)
#define HAVE_ASYNC 1
#endif

typedef __attribute__((address_space(1))) v4i gv4i;
typedef __attribute__((address_space(3))) v4i lv4i;

static __device__ __forceinline__ void async_cp16(void* lds, const void* g) {
#if defined(HAVE_ASYNC)
    __builtin_amdgcn_global_load_async_to_lds_b128(
        (gv4i*)(uintptr_t)g, (lv4i*)(uint32_t)(uintptr_t)lds, 0, 0);
#else
    *(v4i*)lds = *(const v4i*)g;
#endif
}

#if __has_builtin(__builtin_amdgcn_s_wait_asynccnt)
#define WAIT_ASYNC(n) __builtin_amdgcn_s_wait_asynccnt(n)
#elif defined(HAVE_ASYNC)
#define WAIT_ASYNC(n) asm volatile("s_wait_asynccnt %0" ::"i"(n) : "memory")
#else
#define WAIT_ASYNC(n) do { } while (0)
#endif

// ---------------------------------------------------------------------------
// Kernel 0: convert the four 256x256 fp32 weights to fp16 once.
// ---------------------------------------------------------------------------
__global__ __launch_bounds__(256) void convert_w_kernel(
    const float* __restrict__ wq, const float* __restrict__ wk,
    const float* __restrict__ wv, const float* __restrict__ wo,
    half_t* __restrict__ Wh)
{
    const int idx = blockIdx.x * 256 + threadIdx.x;   // grid 256 -> 65536 ids
    Wh[idx]              = (half_t)wq[idx];
    Wh[idx + 1 * 65536]  = (half_t)wk[idx];
    Wh[idx + 2 * 65536]  = (half_t)wv[idx];
    Wh[idx + 3 * 65536]  = (half_t)wo[idx];
}

// ---------------------------------------------------------------------------
// Kernel 1: Q/K/V 1x1-conv projections.  One wave = 16 pixels (one n-tile).
//   Qt, Kt stored [b][n][c] (fp16, Q pre-scaled by 1/16), Vc stored [b][c][n].
// ---------------------------------------------------------------------------
__global__ __launch_bounds__(128) void proj_qkv_kernel(
    const float* __restrict__ x, const half_t* __restrict__ Wh,
    half_t* __restrict__ Qt, half_t* __restrict__ Kt, half_t* __restrict__ Vc)
{
    const int lane = threadIdx.x & 31;
    const int wave = threadIdx.x >> 5;
    const int tile = blockIdx.x * 4 + wave;   // 0..2047
    const int b    = tile >> 8;
    const int n0   = (tile & 255) << 4;
    const int col  = lane & 15;
    const int hid  = lane >> 4;

    const float* xb = x + (size_t)b * CH * NPIX;

    // A fragments of x^T: row n = n0+col (half-wave-coalesced 64B per element).
    v16h xa[8];
    #pragma unroll
    for (int kk = 0; kk < 8; ++kk) {
        #pragma unroll
        for (int e = 0; e < 16; ++e) {
            const int c = kk * 32 + ((e >> 3) << 4) + hid * 8 + (e & 7);
            xa[kk][e] = (half_t)xb[c * NPIX + n0 + col];
        }
    }

    #pragma unroll
    for (int widx = 0; widx < 3; ++widx) {
        const half_t* W = Wh + (size_t)widx * 65536;   // fp16 [o][c]
        for (int ot = 0; ot < 16; ++ot) {
            const int o = ot * 16 + col;               // B-frag column
            v8f acc = zero_v8f();
            #pragma unroll
            for (int kk = 0; kk < 8; ++kk) {
                const int cb = kk * 32 + hid * 8;      // B-frag K block
                v16h bw = frag_from2(W + o * CH + cb, W + o * CH + cb + 16);
                acc = wmma_f16(xa[kk], bw, acc);
            }
            // D layout: row n = n0 + r + 8*hid, col o
            if (widx == 0) {
                #pragma unroll
                for (int r = 0; r < 8; ++r)
                    Qt[(size_t)b * NPIX * CH + (size_t)(n0 + r + hid * 8) * CH +
                       ot * 16 + col] = (half_t)(acc[r] * 0.0625f);  // 1/sqrt(256)
            } else if (widx == 1) {
                #pragma unroll
                for (int r = 0; r < 8; ++r)
                    Kt[(size_t)b * NPIX * CH + (size_t)(n0 + r + hid * 8) * CH +
                       ot * 16 + col] = (half_t)acc[r];
            } else {
                #pragma unroll
                for (int r = 0; r < 8; ++r)
                    Vc[(size_t)b * CH * NPIX + (size_t)(ot * 16 + col) * NPIX +
                       n0 + r + hid * 8] = (half_t)acc[r];
            }
        }
    }
}

// ---------------------------------------------------------------------------
// Kernel 2: flash attention.  WG = 4 waves = 64 queries of one batch.
//   K/V kv-tiles staged in double-buffered LDS via async copies and shared by
//   all 4 waves; each wave owns a 16-query tile and full 256-dim output.
// ---------------------------------------------------------------------------
__global__ __launch_bounds__(128) void flash_attn_kernel(
    const half_t* __restrict__ Qt, const half_t* __restrict__ Kt,
    const half_t* __restrict__ Vc, half_t* __restrict__ AO)
{
    __shared__ half_t Kbuf[2][32 * KSTRIDE];    // 2 x 16.5 KB
    __shared__ half_t Vbuf[2][256 * VSTRIDE];   // 2 x 20   KB
    __shared__ half_t Plds[4][16][40];          // per-wave P transpose tile

    const int tid  = threadIdx.x;
    const int lane = tid & 31;
    const int wave = tid >> 5;
    const int b    = blockIdx.x >> 6;                       // 64 WGs per batch
    const int i0   = (((blockIdx.x & 63) << 2) + wave) << 4;
    const int col  = lane & 15;
    const int hid  = lane >> 4;

    const half_t* Qb = Qt + (size_t)b * NPIX * CH;
    const half_t* Kb = Kt + (size_t)b * NPIX * CH;
    const half_t* Vb = Vc + (size_t)b * CH * NPIX;

    // Resident Q A-fragments (16 queries x 256 channels, scale pre-folded).
    v16h qf[8];
    #pragma unroll
    for (int kk = 0; kk < 8; ++kk) {
        const half_t* qrow = Qb + (size_t)(i0 + col) * CH + kk * 32;
        qf[kk] = frag_from2(qrow + hid * 8, qrow + 16 + hid * 8);
    }

    float m_run[8], l_run[8];
    #pragma unroll
    for (int r = 0; r < 8; ++r) { m_run[r] = -1e30f; l_run[r] = 0.f; }
    v8f oacc[16];
    #pragma unroll
    for (int ct = 0; ct < 16; ++ct) oacc[ct] = zero_v8f();

    // Cooperative stage of one 32-kv block: K rows j0..j0+31 (contiguous 16KB)
    // and V columns j0..j0+31 of all 256 channel rows.  16 x 16B per thread.
    const int sj  = tid >> 2;          // 0..31  (K row / V row group)
    const int sub = tid & 3;           // 16B chunk within 64B
    auto stage = [&](half_t* kb, half_t* vb, int j0) {
        const half_t* Kg = Kb + (size_t)j0 * CH;
        #pragma unroll
        for (int cc = 0; cc < 8; ++cc) {
            const int off = sub * 128 + cc * 16;   // bytes within 512B row
            async_cp16((char*)(kb + sj * KSTRIDE) + off,
                       (const char*)(Kg + (size_t)sj * CH) + off);
        }
        #pragma unroll
        for (int rr = 0; rr < 8; ++rr) {
            const int c = sj + rr * 32;            // V channel row
            async_cp16((char*)(vb + c * VSTRIDE) + sub * 16,
                       (const char*)(Vb + (size_t)c * NPIX + j0) + sub * 16);
        }
    };

    stage(Kbuf[0], Vbuf[0], 0);
    int cur = 0;

    for (int j0 = 0; j0 < NPIX; j0 += 32) {
        if (j0 + 32 < NPIX) {
            stage(Kbuf[cur ^ 1], Vbuf[cur ^ 1], j0 + 32);  // prefetch next
            WAIT_ASYNC(16);       // current tile's 16 ops done; next in flight
        } else {
            WAIT_ASYNC(0);
        }
        __syncthreads();          // staged data visible to all 4 waves

        const half_t* kb = Kbuf[cur];
        const half_t* vb = Vbuf[cur];

        // ---- S = Q^T K for 16 queries x 32 keys (two 16x16 D tiles) ----
        v8f s0 = zero_v8f(), s1 = zero_v8f();
        #pragma unroll
        for (int kk = 0; kk < 8; ++kk) {
            const int cb = kk * 32 + hid * 8;
            const half_t* r0 = kb + (size_t)col * KSTRIDE + cb;
            const half_t* r1 = kb + (size_t)(col + 16) * KSTRIDE + cb;
            s0 = wmma_f16(qf[kk], frag_from2(r0, r0 + 16), s0);
            s1 = wmma_f16(qf[kk], frag_from2(r1, r1 + 16), s1);
        }

        // ---- online softmax (row r + 8*hid lives on one 16-lane group) ----
        float alpha[8];
        #pragma unroll
        for (int r = 0; r < 8; ++r) {
            float t = fmaxf(s0[r], s1[r]);
            t = fmaxf(t, __shfl_xor(t, 1, 32));
            t = fmaxf(t, __shfl_xor(t, 2, 32));
            t = fmaxf(t, __shfl_xor(t, 4, 32));
            t = fmaxf(t, __shfl_xor(t, 8, 32));
            const float m_new = fmaxf(m_run[r], t);
            const float a  = __expf(m_run[r] - m_new);
            const float p0 = __expf(s0[r] - m_new);
            const float p1 = __expf(s1[r] - m_new);
            float rs = p0 + p1;
            rs += __shfl_xor(rs, 1, 32);
            rs += __shfl_xor(rs, 2, 32);
            rs += __shfl_xor(rs, 4, 32);
            rs += __shfl_xor(rs, 8, 32);
            l_run[r] = l_run[r] * a + rs;
            m_run[r] = m_new;
            alpha[r] = a;
            s0[r] = p0; s1[r] = p1;
        }
        #pragma unroll
        for (int ct = 0; ct < 16; ++ct) {
            #pragma unroll
            for (int r = 0; r < 8; ++r) oacc[ct][r] *= alpha[r];
        }

        // ---- D-layout P -> LDS -> A-layout P fragment (wave-private) ----
        #pragma unroll
        for (int r = 0; r < 8; ++r) {
            Plds[wave][r + hid * 8][col]      = (half_t)s0[r];
            Plds[wave][r + hid * 8][16 + col] = (half_t)s1[r];
        }
        asm volatile("s_wait_dscnt 0x0" ::: "memory");
        const half_t* prow = &Plds[wave][col][0];
        v16h pa = frag_from2(prow + hid * 8, prow + 16 + hid * 8);

        // ---- O += P * V^T  (B-frags from shared LDS V tile) ----
        #pragma unroll
        for (int ct = 0; ct < 16; ++ct) {
            const half_t* vr = vb + (size_t)(ct * 16 + col) * VSTRIDE + hid * 8;
            oacc[ct] = wmma_f16(pa, frag_from2(vr, vr + 16), oacc[ct]);
        }

        __syncthreads();          // all waves done reading before overwrite
        cur ^= 1;
    }

    // Normalize and store attention output as fp16 [b][n][c].
    #pragma unroll
    for (int ct = 0; ct < 16; ++ct) {
        #pragma unroll
        for (int r = 0; r < 8; ++r) {
            const float v = oacc[ct][r] / l_run[r];
            AO[(size_t)b * NPIX * CH + (size_t)(i0 + r + hid * 8) * CH +
               ct * 16 + col] = (half_t)v;
        }
    }
}

// ---------------------------------------------------------------------------
// Kernel 3: output projection  out[b][o][n] = sum_c wo[o][c] * AO[b][n][c]
// ---------------------------------------------------------------------------
__global__ __launch_bounds__(128) void proj_out_kernel(
    const half_t* __restrict__ AO, const half_t* __restrict__ woh,
    float* __restrict__ out)
{
    const int lane = threadIdx.x & 31;
    const int wave = threadIdx.x >> 5;
    const int tile = blockIdx.x * 4 + wave;
    const int b    = tile >> 8;
    const int n0   = (tile & 255) << 4;
    const int col  = lane & 15;
    const int hid  = lane >> 4;

    const half_t* Ab = AO + (size_t)b * NPIX * CH;
    v16h af[8];
    #pragma unroll
    for (int kk = 0; kk < 8; ++kk) {
        const half_t* arow = Ab + (size_t)(n0 + col) * CH + kk * 32;
        af[kk] = frag_from2(arow + hid * 8, arow + 16 + hid * 8);
    }

    float* ob = out + (size_t)b * CH * NPIX;
    for (int ot = 0; ot < 16; ++ot) {
        const int o = ot * 16 + col;
        v8f acc = zero_v8f();
        #pragma unroll
        for (int kk = 0; kk < 8; ++kk) {
            const int cb = kk * 32 + hid * 8;
            v16h bw = frag_from2(woh + o * CH + cb, woh + o * CH + cb + 16);
            acc = wmma_f16(af[kk], bw, acc);
        }
        #pragma unroll
        for (int r = 0; r < 8; ++r)
            ob[(size_t)o * NPIX + n0 + r + hid * 8] = acc[r];
    }
}

// ---------------------------------------------------------------------------
extern "C" void kernel_launch(void* const* d_in, const int* in_sizes, int n_in,
                              void* d_out, int out_size, void* d_ws, size_t ws_size,
                              hipStream_t stream) {
    (void)in_sizes; (void)n_in; (void)out_size; (void)ws_size;

    const float* x  = (const float*)d_in[0];
    const float* wq = (const float*)d_in[1];
    const float* wk = (const float*)d_in[2];
    const float* wv = (const float*)d_in[3];
    const float* wo = (const float*)d_in[4];
    float* out = (float*)d_out;

    // Workspace: Qt, Kt ([b][n][c]), Vc ([b][c][n]), AO ([b][n][c]), Wh — fp16.
    const size_t plane = (size_t)BATCH * NPIX * CH;   // 16 MB each (halfs)
    half_t* Qt = (half_t*)d_ws;
    half_t* Kt = Qt + plane;
    half_t* Vc = Kt + plane;
    half_t* AO = Vc + plane;
    half_t* Wh = AO + plane;                          // 4 x 65536 halfs

    const int tiles = BATCH * (NPIX / 16);   // 2048 16-row tiles
    dim3 blk(128);                            // 4 waves of 32
    dim3 grd(tiles / 4);                      // 512 workgroups

    hipLaunchKernelGGL(convert_w_kernel, dim3(256), dim3(256), 0, stream,
                       wq, wk, wv, wo, Wh);
    hipLaunchKernelGGL(proj_qkv_kernel,  grd, blk, 0, stream, x, Wh, Qt, Kt, Vc);
    hipLaunchKernelGGL(flash_attn_kernel, grd, blk, 0, stream, Qt, Kt, Vc, AO);
    hipLaunchKernelGGL(proj_out_kernel,  grd, blk, 0, stream,
                       AO, Wh + 3 * 65536, out);
}
